// RNNRBM_64613488001654
// MI455X (gfx1250) — matrix operand order repaired and checked by reading, not compiled, not run on hardware
//
#include <hip/hip_runtime.h>
#include <hip/hip_bf16.h>

// ---------------------------------------------------------------------------
// RNN-RBM for MI455X (gfx1250, wave32, WMMA 16x16x32 f16->f32)
// T=4096, V=88(->96 padded), H=512, R=512, GIBBS=20
//
// B-fragments (lane = column, 8 consecutive K values plus 8 more at +16 per lane) are loaded as
// two contiguous 16-byte vectors from weights kept in their natural (N x K)
// row-major orientation -- no strided gathers anywhere in the WMMA loops.
// ---------------------------------------------------------------------------

typedef __attribute__((ext_vector_type(16))) _Float16 v16h;
typedef __attribute__((ext_vector_type(8)))  _Float16 v8h;
typedef __attribute__((ext_vector_type(8)))  float    v8f;

union HFrag { v16h v; v8h p[2]; };

#define TCNT   4096
#define TM1    4095   // rows of v_seq
#define VDIM   88
#define VPAD   96
#define HDIM   512
#define RDIM   512
#define GIBBS  20

static __device__ __forceinline__ float sigm(float x) {
  return 1.0f / (1.0f + __expf(-x));
}

// Same-wave LDS ordering fence: HW executes a wave's DS ops in order; we only
// need to stop the compiler from reordering around the cross-lane LDS shuffle.
static __device__ __forceinline__ void wave_lds_fence() {
#if __has_builtin(__builtin_amdgcn_wave_barrier)
  __builtin_amdgcn_wave_barrier();
#else
  __syncthreads();
#endif
  asm volatile("" ::: "memory");
}

static __device__ __forceinline__ v8f wmma_f16(v16h a, v16h b, v8f c) {
  return __builtin_amdgcn_wmma_f32_16x16x32_f16(false, a, false, b,
                                                (short)0, c, false, false);
}

// A-fragment: lane l = row (l&15); halfs 0..7 = K kb+koff8.., halfs 8..15 at +16
static __device__ __forceinline__ v16h load_afrag(const _Float16* __restrict__ base,
                                                  size_t rowstride, int row,
                                                  int kb, int koff8) {
  HFrag a;
  const _Float16* ap = base + (size_t)row * rowstride + kb + koff8;
  a.p[0] = *(const v8h*)ap;
  a.p[1] = *(const v8h*)(ap + 16);
  return a.v;
}

// B-fragment: lane l = col (l&15); halfs 0..7 = K kb+koff16/2.., halfs 8..15 at +16,
// read from an (N x K) row-major buffer.
static __device__ __forceinline__ v16h load_bfrag(const _Float16* __restrict__ base,
                                                  size_t rowstride, int col,
                                                  int kb, int koff16) {
  HFrag b;
  const _Float16* bp = base + (size_t)col * rowstride + kb + (koff16 >> 1);
  b.p[0] = *(const v8h*)bp;
  b.p[1] = *(const v8h*)(bp + 16);
  return b.v;
}

// ---------------------------------------------------------------------------
// Prep: f16 copies of weights in natural (N x K) orientation (padded), one
// real transpose (w^T) for the Gibbs h@w matmul, recurrence init, sync reset.
// ---------------------------------------------------------------------------
__global__ void k_prep(const float* __restrict__ visible,
                       const float* __restrict__ w,
                       const float* __restrict__ wuv,
                       const float* __restrict__ wuh,
                       const float* __restrict__ wvu,
                       const float* __restrict__ bv,
                       const float* __restrict__ u0,
                       _Float16* __restrict__ Vh,     // 4096 x 96  (v_seq)
                       _Float16* __restrict__ Bt_vu,  // 512 x 96   (= wvu pad)
                       _Float16* __restrict__ Bt_uh,  // 512 x 512  (= wuh)
                       _Float16* __restrict__ Bt_uv,  // 96 x 512   (= wuv pad)
                       _Float16* __restrict__ Wp,     // 512 x 96   (= w pad)
                       _Float16* __restrict__ WpT,    // 96 x 512   (= w^T pad)
                       _Float16* __restrict__ Uh,     // 4096 x 512 (u_prev)
                       float* __restrict__ bv_pad,
                       float* __restrict__ ubuf,
                       unsigned* __restrict__ syncv) {
  const int stride = gridDim.x * blockDim.x;
  const int tid0 = blockIdx.x * blockDim.x + threadIdx.x;

  for (int i = tid0; i < TCNT * VPAD; i += stride) {
    int t = i / VPAD, j = i % VPAD;
    float v = (t < TM1 && j < VDIM) ? visible[(size_t)(t + 1) * VDIM + j] : 0.f;
    Vh[i] = (_Float16)v;
  }
  for (int i = tid0; i < RDIM * VPAD; i += stride) {     // wvu (512 x 88) -> pad
    int n = i / VPAD, k = i % VPAD;
    Bt_vu[i] = (_Float16)((k < VDIM) ? wvu[(size_t)n * VDIM + k] : 0.f);
  }
  for (int i = tid0; i < HDIM * RDIM; i += stride)       // wuh (512 x 512)
    Bt_uh[i] = (_Float16)wuh[i];
  for (int i = tid0; i < VPAD * RDIM; i += stride) {     // wuv (88 x 512) -> pad rows
    int n = i / RDIM, k = i % RDIM;
    Bt_uv[i] = (_Float16)((n < VDIM) ? wuv[(size_t)n * RDIM + k] : 0.f);
  }
  for (int i = tid0; i < HDIM * VPAD; i += stride) {     // w (512 x 88) -> pad
    int h = i / VPAD, j = i % VPAD;
    Wp[i] = (_Float16)((j < VDIM) ? w[(size_t)h * VDIM + j] : 0.f);
  }
  for (int i = tid0; i < VPAD * HDIM; i += stride) {     // w^T (96 x 512)
    int j = i / HDIM, h = i % HDIM;
    WpT[i] = (_Float16)((j < VDIM) ? w[(size_t)h * VDIM + j] : 0.f);
  }
  for (int i = tid0; i < TCNT * RDIM; i += stride)
    Uh[i] = (_Float16)((i < RDIM) ? u0[i] : 0.f);
  for (int i = tid0; i < VPAD; i += stride) bv_pad[i] = (i < VDIM) ? bv[i] : 0.f;
  for (int i = tid0; i < RDIM; i += stride) ubuf[i] = u0[i];  // ping slot 0
  if (tid0 == 0) *syncv = 0u;  // reset every launch (graph replay safe)
}

// ---------------------------------------------------------------------------
// Regularizer: 0.2 * (||wuv||_F + ||wuh||_F), fixed-order reduction.
// ---------------------------------------------------------------------------
__global__ void __launch_bounds__(256) k_reg(const float* __restrict__ wuv,
                                             const float* __restrict__ wuh,
                                             float* __restrict__ reg_ws,
                                             float* __restrict__ out_reg) {
  __shared__ float red[256];
  const int tid = threadIdx.x;
  float s1 = 0.f, s2 = 0.f;
  for (int i = tid; i < VDIM * RDIM; i += 256) { float x = wuv[i]; s1 += x * x; }
  for (int i = tid; i < HDIM * RDIM; i += 256) { float x = wuh[i]; s2 += x * x; }
  red[tid] = s1; __syncthreads();
  for (int o = 128; o > 0; o >>= 1) { if (tid < o) red[tid] += red[tid + o]; __syncthreads(); }
  float n1 = red[0]; __syncthreads();
  red[tid] = s2; __syncthreads();
  for (int o = 128; o > 0; o >>= 1) { if (tid < o) red[tid] += red[tid + o]; __syncthreads(); }
  if (tid == 0) {
    float r = 0.2f * (sqrtf(n1) + sqrtf(red[0]));
    reg_ws[0] = r;
    out_reg[0] = r;   // d_out[4096]
  }
}

// ---------------------------------------------------------------------------
// f16 WMMA GEMM: C(4096 x N, f32) = A(4096 x K, f16) * Bt^T + bias
// Bt is (N x K) row-major. One 16x16 tile per wave, 8 waves/block, K unrolled.
// ---------------------------------------------------------------------------
template <int K>
__global__ void __launch_bounds__(256) k_gemm(const _Float16* __restrict__ A,
                                              const _Float16* __restrict__ Bt,
                                              const float* __restrict__ bias,
                                              float* __restrict__ C,
                                              int Ntiles, int lda, int ldc) {
  const int lane = threadIdx.x & 31;
  const int wave = threadIdx.x >> 5;
  const int tile = blockIdx.x * 8 + wave;
  const int mt = tile / Ntiles, nt = tile % Ntiles;
  const int l15 = lane & 15;
  const int row = mt * 16 + l15;
  const int col = nt * 16 + l15;
  const int koff8  = (lane >= 16) ? 8 : 0;
  const int koff16 = (lane >= 16) ? 16 : 0;

  v8f acc = {};
#pragma unroll
  for (int kb = 0; kb < K; kb += 32) {
    v16h a = load_afrag(A, lda, row, kb, koff8);
    v16h b = load_bfrag(Bt, K, col, kb, koff16);
    acc = wmma_f16(a, b, acc);
  }
  const float bs = bias[col];
#pragma unroll
  for (int j = 0; j < 8; ++j) {
    int r = mt * 16 + j + ((lane >= 16) ? 8 : 0);
    C[(size_t)r * ldc + col] = acc[j] + bs;
  }
}

// ---------------------------------------------------------------------------
// Sequential recurrence: u_{s+1} = tanh(Wv[s] + wuu @ u_s).
// 8 cooperating blocks x 512 threads; 64 rows/block, 8 threads/row.
// Grid barrier: monotone atomic counter (reset by k_prep each launch).
// ---------------------------------------------------------------------------
__global__ void __launch_bounds__(512) k_recur(const float* __restrict__ wuu,
                                               const float* __restrict__ Wv,
                                               float* __restrict__ ubuf,
                                               _Float16* __restrict__ Uh,
                                               unsigned* __restrict__ syncv) {
  __shared__ float u_s[RDIM];
  const int tid = threadIdx.x;
  const int row = blockIdx.x * 64 + (tid >> 3);
  const int kc = tid & 7;
  const float* wrow = wuu + (size_t)row * RDIM;

  for (int s = 0; s < TM1 - 1; ++s) {          // 4094 steps
    const float* rd = ubuf + (size_t)(s & 1) * RDIM;
    float* wr = ubuf + (size_t)((s + 1) & 1) * RDIM;
    u_s[tid] = rd[tid];
    __syncthreads();
    float p = 0.f;
#pragma unroll 8
    for (int i = 0; i < 64; ++i) {
      int k = i * 8 + kc;                      // coalesced across the 8 lanes
      p += wrow[k] * u_s[k];
    }
    p += __shfl_xor(p, 1, 32);
    p += __shfl_xor(p, 2, 32);
    p += __shfl_xor(p, 4, 32);
    if (kc == 0) {
      float un = tanhf(p + Wv[(size_t)s * RDIM + row]);
      wr[row] = un;
      Uh[(size_t)(s + 1) * RDIM + row] = (_Float16)un;
    }
    // grid-wide barrier among 8 resident blocks
    __threadfence();
    __syncthreads();
    if (tid == 0) {
      atomicAdd(syncv, 1u);
      const unsigned target = 8u * (unsigned)(s + 1);
      while (__hip_atomic_load(syncv, __ATOMIC_ACQUIRE,
                               __HIP_MEMORY_SCOPE_AGENT) < target)
        __builtin_amdgcn_s_sleep(1);
    }
    __syncthreads();
  }
}

// ---------------------------------------------------------------------------
// Gibbs sampler: 1 wave owns a 16-row tile for all 20 steps (rows independent).
// Per step:  p_h = sigm(v @ w^T + bh_t)  -> h binary (h = p_h > rand)
//            p_v = sigm(h @ w   + bv_t)  -> v binary
// B-frags come from Wp (512x96) and WpT (96x512), always contiguous 16B pairs.
// Transposes between C-layout and A-frag layout via per-wave LDS buffers.
// Also emits per-row mse into d_out[1..4095].
// ---------------------------------------------------------------------------
__global__ void __launch_bounds__(256) k_gibbs(const _Float16* __restrict__ Vh,
                                               const _Float16* __restrict__ Wp,
                                               const _Float16* __restrict__ WpT,
                                               const float* __restrict__ BH,
                                               const float* __restrict__ BV,
                                               const float* __restrict__ rand_h,
                                               const float* __restrict__ rand_v,
                                               float* __restrict__ d_out) {
  __shared__ _Float16 hbuf_s[8 * 16 * 32];   //  8 KB: per-wave h chunk (16x32)
  __shared__ _Float16 vbuf_s[8 * 16 * 96];   // 24 KB: per-wave v tile  (16x96)
  const int lane = threadIdx.x & 31;
  const int wave = threadIdx.x >> 5;
  _Float16* hb = hbuf_s + wave * (16 * 32);
  _Float16* vb = vbuf_s + wave * (16 * 96);
  const int tile = blockIdx.x * 8 + wave;
  const int t0 = tile * 16;
  const int l15 = lane & 15;
  const int koff8  = (lane >= 16) ? 8 : 0;
  const int koff16 = (lane >= 16) ? 16 : 0;
  const int rowoff = (lane >= 16) ? 8 : 0;
  const v8f zero = {};

  // initial v fragments (A-layout) straight from packed global v_seq
  v16h vfrag[3];
#pragma unroll
  for (int kb = 0; kb < 3; ++kb)
    vfrag[kb] = load_afrag(Vh, VPAD, t0 + l15, kb * 32, koff8);

  for (int g = 0; g < GIBBS; ++g) {
    if (g + 1 < GIBBS)   // warm L2 for the next step's random stream
      __builtin_prefetch(rand_h + ((size_t)t0 * GIBBS + g + 1) * HDIM, 0, 1);

    v8f pv[6];
#pragma unroll
    for (int i = 0; i < 6; ++i) pv[i] = zero;

    for (int kc = 0; kc < 16; ++kc) {          // 16 chunks of 32 h-columns
#pragma unroll
      for (int sub = 0; sub < 2; ++sub) {
        const int hn = (kc * 2 + sub) * 16 + l15;   // h column (0..511)
        v8f acc = zero;
#pragma unroll
        for (int kb = 0; kb < 3; ++kb)              // K over V (96)
          acc = wmma_f16(vfrag[kb],
                         load_bfrag(Wp, VPAD, hn, kb * 32, koff16), acc);
#pragma unroll
        for (int j = 0; j < 8; ++j) {
          const int trow = t0 + j + rowoff;
          float h = 0.f;
          if (trow < TM1) {
            float x = acc[j] + BH[(size_t)trow * HDIM + hn];
            float r = rand_h[((size_t)trow * GIBBS + g) * HDIM + hn];
            h = (sigm(x) > r) ? 1.f : 0.f;
          }
          hb[(j + rowoff) * 32 + sub * 16 + l15] = (_Float16)h;
        }
      }
      wave_lds_fence();
      // h chunk as A-fragment (transpose through LDS)
      v16h hf = load_afrag(hb, 32, l15, 0, koff8);
#pragma unroll
      for (int n2 = 0; n2 < 6; ++n2)                // p_v += h_chunk @ w_chunk
        pv[n2] = wmma_f16(hf,
                          load_bfrag(WpT, HDIM, n2 * 16 + l15, kc * 32, koff16),
                          pv[n2]);
      wave_lds_fence();   // keep reads of hb ordered before next chunk's writes
    }

    // sample v, stash row-major tile in LDS, reload as A-fragments
#pragma unroll
    for (int n2 = 0; n2 < 6; ++n2) {
      const int vcol = n2 * 16 + l15;
#pragma unroll
      for (int j = 0; j < 8; ++j) {
        const int trow = t0 + j + rowoff;
        float vn = 0.f;
        if (trow < TM1 && vcol < VDIM) {
          float x = pv[n2][j] + BV[(size_t)trow * VPAD + vcol];
          float r = rand_v[((size_t)trow * GIBBS + g) * VDIM + vcol];
          vn = (sigm(x) > r) ? 1.f : 0.f;
        }
        vb[(j + rowoff) * VPAD + vcol] = (_Float16)vn;
      }
    }
    wave_lds_fence();
#pragma unroll
    for (int kb = 0; kb < 3; ++kb)
      vfrag[kb] = load_afrag(vb, VPAD, l15, kb * 32, koff8);
    wave_lds_fence();
  }

  // mse[t] = mean_j |v_seq - neg| ; 2 lanes per row, 44 columns each
  const int rr = lane >> 1;
  const int half = lane & 1;
  const int trow = t0 + rr;
  float s = 0.f;
  for (int c = half * 44; c < half * 44 + 44; ++c) {
    float nv = (float)vb[rr * VPAD + c];
    float ov = (float)Vh[(size_t)trow * VPAD + c];
    s += fabsf(ov - nv);
  }
  s += __shfl_xor(s, 1, 32);
  if (half == 0 && trow < TM1)
    d_out[1 + trow] = s * (1.f / (float)VDIM);
}

// ---------------------------------------------------------------------------
// Cross-entropy: fixed-order two-stage reduction (deterministic).
// ---------------------------------------------------------------------------
__global__ void __launch_bounds__(256) k_ce(const float* __restrict__ visible,
                                            const float* __restrict__ BV,
                                            float* __restrict__ partials) {
  __shared__ float red[256];
  const int tid = threadIdx.x;
  const int N = TM1 * VDIM;
  float s = 0.f;
  for (int i = blockIdx.x * 256 + tid; i < N; i += 256 * 256) {
    int t = i / VDIM, j = i % VDIM;
    float y = sigm(BV[(size_t)t * VPAD + j]);
    float v = visible[(size_t)(t + 1) * VDIM + j];
    s += -(v * logf(1e-6f + y) + (1.f - v) * logf(1e-6f + 1.f - y));
  }
  red[tid] = s; __syncthreads();
  for (int o = 128; o > 0; o >>= 1) { if (tid < o) red[tid] += red[tid + o]; __syncthreads(); }
  if (tid == 0) partials[blockIdx.x] = red[0];
}

__global__ void k_final(const float* __restrict__ partials,
                        const float* __restrict__ reg_ws,
                        float* __restrict__ d_out) {
  if (threadIdx.x == 0 && blockIdx.x == 0) {
    float s = 0.f;
    for (int i = 0; i < 256; ++i) s += partials[i];
    d_out[0] = s * (1.f / (float)TCNT) + reg_ws[0];
  }
}

// ---------------------------------------------------------------------------
extern "C" void kernel_launch(void* const* d_in, const int* in_sizes, int n_in,
                              void* d_out, int out_size, void* d_ws, size_t ws_size,
                              hipStream_t stream) {
  const float* visible = (const float*)d_in[0];
  const float* rand_h  = (const float*)d_in[1];
  const float* rand_v  = (const float*)d_in[2];
  const float* w       = (const float*)d_in[3];
  const float* wuu     = (const float*)d_in[4];
  const float* wuv     = (const float*)d_in[5];
  const float* wuh     = (const float*)d_in[6];
  const float* wvu     = (const float*)d_in[7];
  const float* bv      = (const float*)d_in[8];
  const float* bh      = (const float*)d_in[9];
  const float* bu      = (const float*)d_in[10];
  const float* u0      = (const float*)d_in[11];
  float* out = (float*)d_out;

  char* p = (char*)d_ws;
  auto alloc = [&](size_t bytes) -> void* {
    void* r = (void*)p;
    p += (bytes + 255) & ~(size_t)255;
    return r;
  };
  _Float16* Vh     = (_Float16*)alloc((size_t)TCNT * VPAD * 2);   // v_seq f16 padded
  _Float16* Bt_vu  = (_Float16*)alloc((size_t)RDIM * VPAD * 2);
  _Float16* Bt_uh  = (_Float16*)alloc((size_t)HDIM * RDIM * 2);
  _Float16* Bt_uv  = (_Float16*)alloc((size_t)VPAD * RDIM * 2);
  _Float16* Wp     = (_Float16*)alloc((size_t)HDIM * VPAD * 2);
  _Float16* WpT    = (_Float16*)alloc((size_t)VPAD * HDIM * 2);
  _Float16* Uh     = (_Float16*)alloc((size_t)TCNT * RDIM * 2);
  float*    Wv     = (float*)alloc((size_t)TCNT * RDIM * 4);
  float*    BH     = (float*)alloc((size_t)TCNT * HDIM * 4);
  float*    BV     = (float*)alloc((size_t)TCNT * VPAD * 4);
  float*    bv_pad = (float*)alloc(VPAD * 4);
  float*    ubuf   = (float*)alloc(2 * RDIM * 4);
  float*    parts  = (float*)alloc(256 * 4);
  float*    reg_ws = (float*)alloc(4);
  unsigned* syncv  = (unsigned*)alloc(4);

  // 1. pack operands + reset state
  k_prep<<<512, 256, 0, stream>>>(visible, w, wuv, wuh, wvu, bv, u0,
                                  Vh, Bt_vu, Bt_uh, Bt_uv, Wp, WpT, Uh,
                                  bv_pad, ubuf, syncv);
  // 2. regularizer -> d_out[4096] and ws
  k_reg<<<1, 256, 0, stream>>>(wuv, wuh, reg_ws, out + 4096);
  // 3. Wv = v_seq @ wvu^T + bu   (M=4096, N=512, K=96)
  k_gemm<VPAD><<<1024, 256, 0, stream>>>(Vh, Bt_vu, bu, Wv, 32, VPAD, RDIM);
  // 4. sequential recurrence -> Uh (u_prev, f16)
  k_recur<<<8, 512, 0, stream>>>(wuu, Wv, ubuf, Uh, syncv);
  // 5. BH = u_prev @ wuh^T + bh  (M=4096, N=512, K=512)
  k_gemm<RDIM><<<1024, 256, 0, stream>>>(Uh, Bt_uh, bh, BH, 32, RDIM, HDIM);
  // 6. BV = u_prev @ wuv^T + bv  (M=4096, N=96, K=512)
  k_gemm<RDIM><<<192, 256, 0, stream>>>(Uh, Bt_uv, bv_pad, BV, 6, RDIM, VPAD);
  // 7. 20 Gibbs steps + mse -> d_out[1..4095]
  k_gibbs<<<32, 256, 0, stream>>>(Vh, Wp, WpT, BH, BV, rand_h, rand_v, out);
  // 8-9. cross-entropy -> d_out[0]
  k_ce<<<256, 256, 0, stream>>>(visible, BV, parts);
  k_final<<<1, 1, 0, stream>>>(parts, reg_ws, out);
}
